// Head_75204877353135
// MI455X (gfx1250) — matrix operand order repaired and checked by reading, not compiled, not run on hardware
//
#include <hip/hip_runtime.h>
#include <hip/hip_bf16.h>

typedef __attribute__((ext_vector_type(16))) __bf16 v16bf;
typedef __attribute__((ext_vector_type(8)))  float  v8f;

#define NROWS 8192
#define FIN   512
#define DHEAD 64

// A-fragment (16x32, 16-bit) per-lane K offset, contiguous order (matches B):
// lanes 0-15: K 0..15
// lanes16-31: K 16..31
__device__ __forceinline__ int a_koff(int j, int hf) {
  return (hf << 4) + j;
}

// ---------------------------------------------------------------------------
// Kernel 1: QKV projection. out[r][c] = sum_k X[r][k] * W[c][k], bf16 results.
// Q is pre-scaled by (1/sqrt(64))*log2(e) so the attention loop's exp2-domain
// scores come straight out of the WMMA. Q, K row-major [8192][64]; V stored
// transposed [64][8192] so P@V B-fragments are contiguous loads.
// ---------------------------------------------------------------------------
__global__ __launch_bounds__(32) void qkv_proj_kernel(
    const float* __restrict__ X,
    const float* __restrict__ Wq, const float* __restrict__ Wk,
    const float* __restrict__ Wv,
    __bf16* __restrict__ Qb, __bf16* __restrict__ Kb, __bf16* __restrict__ Vt) {
  const int rowBase = blockIdx.x * 16;
  const int which   = blockIdx.y;            // 0:Q 1:K 2:V
  const float* W = (which == 0) ? Wq : (which == 1) ? Wk : Wv;
  const int lane = threadIdx.x;
  const int hf = lane >> 4;
  const int lm = lane & 15;

  v8f acc[4] = {};
  const float* xrow = X + (size_t)(rowBase + lm) * FIN;

  for (int kb = 0; kb < FIN; kb += 32) {
    v16bf a;
#pragma unroll
    for (int j = 0; j < 16; ++j) a[j] = (__bf16)xrow[kb + a_koff(j, hf)];
#pragma unroll
    for (int nt = 0; nt < 4; ++nt) {
      const float* wrow = W + (size_t)(nt * 16 + lm) * FIN + kb + hf * 16;
      v16bf b;
#pragma unroll
      for (int j = 0; j < 16; ++j) b[j] = (__bf16)wrow[j];
      acc[nt] = __builtin_amdgcn_wmma_f32_16x16x32_bf16(
          false, a, false, b, (short)0, acc[nt], false, false);
    }
  }

  const float qscale = (which == 0) ? 0.125f * 1.44269504088896f : 1.0f;
  if (which == 2) {
#pragma unroll
    for (int nt = 0; nt < 4; ++nt)
#pragma unroll
      for (int v = 0; v < 8; ++v)
        Vt[(size_t)(nt * 16 + lm) * NROWS + rowBase + v + hf * 8] =
            (__bf16)acc[nt][v];
  } else {
    __bf16* out = (which == 0) ? Qb : Kb;
#pragma unroll
    for (int nt = 0; nt < 4; ++nt)
#pragma unroll
      for (int v = 0; v < 8; ++v)
        out[(size_t)(rowBase + v + hf * 8) * DHEAD + nt * 16 + lm] =
            (__bf16)(acc[nt][v] * qscale);
  }
}

// ---------------------------------------------------------------------------
// Kernel 2: flash attention with int32 mask. One wave = 16 query rows,
// 64 keys per iteration. Mask (256 MB, read-once) streamed with NT hints so
// it never evicts the 2 MB bf16 K/V working set from the 192 MB L2.
// Row stats live in the WMMA C-fragment layout (VGPR v <-> row v+hf*8,
// lanes 0-15/16-31 <-> columns) so reductions are 4-stage 16-lane shfl_xor.
// Row max is taken over raw (mask-included) scores -- softmax is invariant
// to any m >= true max and scores are O(1) -- mask applied as p = m?exp2:0.
// ---------------------------------------------------------------------------
__global__ __launch_bounds__(32) void flash_attn_kernel(
    const __bf16* __restrict__ Qb, const __bf16* __restrict__ Kb,
    const __bf16* __restrict__ Vt, const int* __restrict__ mask,
    float* __restrict__ H) {
  __shared__ __bf16 pLds[16 * 64];           // P tile, row-major [16][64]
  const int qBase = blockIdx.x * 16;
  const int lane = threadIdx.x;
  const int hf = lane >> 4;
  const int lm = lane & 15;

  // Q A-fragments (pre-scaled into exp2 domain by kernel 1)
  v16bf qa[2];
  const __bf16* qrow = Qb + (size_t)(qBase + lm) * DHEAD;
#pragma unroll
  for (int ks = 0; ks < 2; ++ks)
#pragma unroll
    for (int j = 0; j < 16; ++j) qa[ks][j] = qrow[ks * 32 + a_koff(j, hf)];

  v8f o[4] = {};
  float Mrow[8], Lrow[8];
#pragma unroll
  for (int v = 0; v < 8; ++v) { Mrow[v] = -1e30f; Lrow[v] = 0.f; }

  for (int kb = 0; kb < NROWS; kb += 64) {
    // ---- S = Q @ K^T for a 16x64 tile (already in exp2 domain) ----
    v8f s[4] = {};
#pragma unroll
    for (int ks = 0; ks < 2; ++ks)
#pragma unroll
      for (int nt = 0; nt < 4; ++nt) {
        v16bf b = *(const v16bf*)(Kb + (size_t)(kb + nt * 16 + lm) * DHEAD +
                                  ks * 32 + hf * 16);
        s[nt] = __builtin_amdgcn_wmma_f32_16x16x32_bf16(
            false, qa[ks], false, b, (short)0, s[nt], false, false);
      }

    // ---- mask + online softmax ----
    float corr[8];
#pragma unroll
    for (int v = 0; v < 8; ++v) {
      const int* mp = mask + (size_t)(qBase + v + hf * 8) * NROWS + kb + lm;
      int m0 = __builtin_nontemporal_load(mp);
      int m1 = __builtin_nontemporal_load(mp + 16);
      int m2 = __builtin_nontemporal_load(mp + 32);
      int m3 = __builtin_nontemporal_load(mp + 48);
      float e0 = s[0][v], e1 = s[1][v], e2 = s[2][v], e3 = s[3][v];
      float t = fmaxf(fmaxf(e0, e1), fmaxf(e2, e3));
      t = fmaxf(t, __shfl_xor(t, 1));
      t = fmaxf(t, __shfl_xor(t, 2));
      t = fmaxf(t, __shfl_xor(t, 4));
      t = fmaxf(t, __shfl_xor(t, 8));      // row max over this 64-key tile
      float mnew = fmaxf(Mrow[v], t);
      corr[v] = exp2f(Mrow[v] - mnew);
      Mrow[v] = mnew;
      float p0 = m0 ? exp2f(e0 - mnew) : 0.f;
      float p1 = m1 ? exp2f(e1 - mnew) : 0.f;
      float p2 = m2 ? exp2f(e2 - mnew) : 0.f;
      float p3 = m3 ? exp2f(e3 - mnew) : 0.f;
      s[0][v] = p0; s[1][v] = p1; s[2][v] = p2; s[3][v] = p3;
      float rs = (p0 + p1) + (p2 + p3);
      rs += __shfl_xor(rs, 1);
      rs += __shfl_xor(rs, 2);
      rs += __shfl_xor(rs, 4);
      rs += __shfl_xor(rs, 8);             // row sum of masked exp
      Lrow[v] = Lrow[v] * corr[v] + rs;
    }

    // rescale running O accumulators
#pragma unroll
    for (int v = 0; v < 8; ++v) {
      o[0][v] *= corr[v]; o[1][v] *= corr[v];
      o[2][v] *= corr[v]; o[3][v] *= corr[v];
    }

    // ---- transpose P: C-layout -> A-fragment layout via LDS ----
#pragma unroll
    for (int nt = 0; nt < 4; ++nt)
#pragma unroll
      for (int v = 0; v < 8; ++v)
        pLds[(v + hf * 8) * 64 + nt * 16 + lm] = (__bf16)s[nt][v];
    __syncthreads();                        // 1-wave WG: lowers to ds waits
    v16bf pa[2];
    const __bf16* prow = &pLds[lm * 64];
#pragma unroll
    for (int ps = 0; ps < 2; ++ps)
#pragma unroll
      for (int j = 0; j < 16; ++j)          // two contiguous 8-elt runs ->
        pa[ps][j] = prow[ps * 32 + a_koff(j, hf)];  // ds_load_b128 pairs
    __syncthreads();

    // ---- O += P(16x64) @ V(64x64) from pre-transposed Vt ----
#pragma unroll
    for (int ps = 0; ps < 2; ++ps)
#pragma unroll
      for (int nt = 0; nt < 4; ++nt) {
        v16bf vb = *(const v16bf*)(Vt + (size_t)(nt * 16 + lm) * NROWS +
                                   kb + ps * 32 + hf * 16);
        o[nt] = __builtin_amdgcn_wmma_f32_16x16x32_bf16(
            false, pa[ps], false, vb, (short)0, o[nt], false, false);
      }
  }

  // ---- finalize: H = O / L ----
#pragma unroll
  for (int v = 0; v < 8; ++v) {
    float inv = 1.0f / Lrow[v];
    float* hrow = H + (size_t)(qBase + v + hf * 8) * DHEAD;
#pragma unroll
    for (int nt = 0; nt < 4; ++nt) hrow[nt * 16 + lm] = o[nt][v] * inv;
  }
}

extern "C" void kernel_launch(void* const* d_in, const int* in_sizes, int n_in,
                              void* d_out, int out_size, void* d_ws, size_t ws_size,
                              hipStream_t stream) {
  (void)in_sizes; (void)n_in; (void)out_size; (void)ws_size;
  const float* X    = (const float*)d_in[0];
  const int*   mask = (const int*)d_in[1];
  const float* Wk   = (const float*)d_in[2];
  const float* Wq   = (const float*)d_in[3];
  const float* Wv   = (const float*)d_in[4];
  float* H = (float*)d_out;

  // workspace: Qb | Kb (row-major [8192][64]) | Vt (transposed [64][8192])
  __bf16* Qb = (__bf16*)d_ws;
  __bf16* Kb = Qb + (size_t)NROWS * DHEAD;
  __bf16* Vt = Kb + (size_t)NROWS * DHEAD;

  qkv_proj_kernel<<<dim3(NROWS / 16, 3), 32, 0, stream>>>(X, Wq, Wk, Wv, Qb, Kb, Vt);
  flash_attn_kernel<<<dim3(NROWS / 16), 32, 0, stream>>>(Qb, Kb, Vt, mask, H);
}
